// DecoderLayer_11132555231785
// MI455X (gfx1250) — matrix operand order repaired and checked by reading, not compiled, not run on hardware
//
#include <hip/hip_runtime.h>
#include <hip/hip_bf16.h>

#define H    128
#define NIN  384
#define KNB  48
#define FF   512
#define CIN  512          // H + NIN
#define SCALEF 30.0f
#define EPSF 1e-5f
#define NNODES 4096       // B*N = 2*2048

typedef __attribute__((ext_vector_type(16))) __bf16 bf16x16;
typedef __attribute__((ext_vector_type(8)))  float  f32x8;

union FragAB {
  bf16x16 v;
  uint4   q[2];
  unsigned short u[16];
};

__device__ __forceinline__ unsigned short f2bf(float x) {
  unsigned int u = __float_as_uint(x);
  u += 0x7FFFu + ((u >> 16) & 1u);   // round-to-nearest-even
  return (unsigned short)(u >> 16);
}

__device__ __forceinline__ float gelu_erf(float x) {
  return 0.5f * x * (1.0f + erff(x * 0.70710678118654752f));
}

// ---------------------------------------------------------------------------
// Weight prep: W [out=NT*16, in=Kdim] f32 row-major  ->  bf16 WMMA B-fragments.
// Layout per (kc, nt): lane l holds column n = nt*16 + (l&15), 16 K values
// k = kc*32 + (l>>4)*8 + (j&7) + (j>>3)*16  (ISA 32x16 bf16 B-operand layout).
// ---------------------------------------------------------------------------
__global__ void prep_frag(const float* __restrict__ W, unsigned short* __restrict__ out,
                          int Kdim, int NT, int total) {
  int eid = blockIdx.x * blockDim.x + threadIdx.x;
  if (eid >= total) return;
  int j    = eid & 15;
  int lane = (eid >> 4) & 31;
  int nt   = (eid >> 9) % NT;
  int kc   = eid / (NT << 9);
  int k = kc * 32 + (lane >> 4) * 8 + (j & 7) + (j >> 3) * 16;
  int n = nt * 16 + (lane & 15);
  out[eid] = f2bf(W[(size_t)n * Kdim + k]);
}

// ---------------------------------------------------------------------------
// Message MLP + K-reduction + LN1, fused per node. 256 threads = 8 waves.
// ---------------------------------------------------------------------------
struct __align__(16) SMemB {
  union {
    unsigned short Xs[KNB * 520];   // bf16 X [48 x 512], padded stride 520
    float          Fs[KNB * 132];   // masked layer-3 output, padded stride 132
  };
  unsigned short Y0[KNB * 136];     // bf16 M1 [48 x 128], padded stride 136
  unsigned short Y1[KNB * 136];     // bf16 M2
  float sV[H];
  float maskA[KNB];
  float Sred[H];
};

__global__ void __launch_bounds__(256)
msg_kernel(const float* __restrict__ hV, const float* __restrict__ hE,
           const float* __restrict__ maskAtt,
           const unsigned short* __restrict__ W1f, const float* __restrict__ b1,
           const unsigned short* __restrict__ W2f, const float* __restrict__ b2,
           const unsigned short* __restrict__ W3f, const float* __restrict__ b3,
           const float* __restrict__ g1, const float* __restrict__ be1,
           float* __restrict__ h1f, unsigned short* __restrict__ h1b)
{
  __shared__ SMemB sm;
  const int tid  = threadIdx.x;
  const int lane = tid & 31;
  const int w    = tid >> 5;
  const int node = blockIdx.x;
  const int col  = lane & 15;
  const int half = lane >> 4;

  if (tid < H)   sm.sV[tid]    = hV[(size_t)node * H + tid];
  if (tid < KNB) sm.maskA[tid] = maskAtt[(size_t)node * KNB + tid];
  __syncthreads();

  // Build X = [h_V || h_E] in bf16, LDS, padded rows (HBM read of h_E happens once here)
  const float* hEbase = hE + (size_t)node * KNB * NIN;
  for (int i4 = tid; i4 < KNB * (CIN / 4); i4 += 256) {
    int row = i4 >> 7;
    int c4  = (i4 & 127) << 2;
    float4 v;
    if (c4 < H) {
      v.x = sm.sV[c4]; v.y = sm.sV[c4 + 1]; v.z = sm.sV[c4 + 2]; v.w = sm.sV[c4 + 3];
    } else {
      v = *(const float4*)(hEbase + (size_t)row * NIN + (c4 - H));
    }
    ushort4 o; o.x = f2bf(v.x); o.y = f2bf(v.y); o.z = f2bf(v.z); o.w = f2bf(v.w);
    *(ushort4*)(&sm.Xs[row * 520 + c4]) = o;
  }
  __syncthreads();

  // ---- Layer 1: X[48,512] @ W1^T -> GELU -> Y0[48,128] ----
  for (int jj = 0; jj < 3; ++jj) {
    int job = w + jj * 8;
    int mt = job >> 3, nt = job & 7;
    f32x8 c = {};
    int arow = mt * 16 + col;
    for (int kc = 0; kc < 16; ++kc) {
      FragAB a, b;
      int abase = arow * 520 + kc * 32 + half * 8;
      a.q[0] = *(const uint4*)(&sm.Xs[abase]);
      a.q[1] = *(const uint4*)(&sm.Xs[abase + 16]);
      const unsigned short* bp = W1f + (((kc * 8 + nt) * 32 + lane) << 4);
      b.q[0] = *(const uint4*)(bp);
      b.q[1] = *(const uint4*)(bp + 8);
      c = __builtin_amdgcn_wmma_f32_16x16x32_bf16(false, a.v, false, b.v, (short)0, c, false, false);
    }
    float bias = b1[nt * 16 + col];
#pragma unroll
    for (int r = 0; r < 8; ++r) {
      int m = mt * 16 + half * 8 + r;
      sm.Y0[m * 136 + nt * 16 + col] = f2bf(gelu_erf(c[r] + bias));
    }
  }
  __syncthreads();

  // ---- Layer 2: Y0[48,128] @ W2^T -> GELU -> Y1[48,128] ----
  for (int jj = 0; jj < 3; ++jj) {
    int job = w + jj * 8;
    int mt = job >> 3, nt = job & 7;
    f32x8 c = {};
    int arow = mt * 16 + col;
    for (int kc = 0; kc < 4; ++kc) {
      FragAB a, b;
      int abase = arow * 136 + kc * 32 + half * 8;
      a.q[0] = *(const uint4*)(&sm.Y0[abase]);
      a.q[1] = *(const uint4*)(&sm.Y0[abase + 16]);
      const unsigned short* bp = W2f + (((kc * 8 + nt) * 32 + lane) << 4);
      b.q[0] = *(const uint4*)(bp);
      b.q[1] = *(const uint4*)(bp + 8);
      c = __builtin_amdgcn_wmma_f32_16x16x32_bf16(false, a.v, false, b.v, (short)0, c, false, false);
    }
    float bias = b2[nt * 16 + col];
#pragma unroll
    for (int r = 0; r < 8; ++r) {
      int m = mt * 16 + half * 8 + r;
      sm.Y1[m * 136 + nt * 16 + col] = f2bf(gelu_erf(c[r] + bias));
    }
  }
  __syncthreads();

  // ---- Layer 3: Y1[48,128] @ W3^T + b3, masked -> Fs (f32) ----
  for (int jj = 0; jj < 3; ++jj) {
    int job = w + jj * 8;
    int mt = job >> 3, nt = job & 7;
    f32x8 c = {};
    int arow = mt * 16 + col;
    for (int kc = 0; kc < 4; ++kc) {
      FragAB a, b;
      int abase = arow * 136 + kc * 32 + half * 8;
      a.q[0] = *(const uint4*)(&sm.Y1[abase]);
      a.q[1] = *(const uint4*)(&sm.Y1[abase + 16]);
      const unsigned short* bp = W3f + (((kc * 8 + nt) * 32 + lane) << 4);
      b.q[0] = *(const uint4*)(bp);
      b.q[1] = *(const uint4*)(bp + 8);
      c = __builtin_amdgcn_wmma_f32_16x16x32_bf16(false, a.v, false, b.v, (short)0, c, false, false);
    }
    int n = nt * 16 + col;
    float bias = b3[n];
#pragma unroll
    for (int r = 0; r < 8; ++r) {
      int m = mt * 16 + half * 8 + r;
      sm.Fs[m * 132 + n] = (c[r] + bias) * sm.maskA[m];
    }
  }
  __syncthreads();

  // ---- sum over K, /SCALE, residual ----
  if (tid < H) {
    float s = 0.f;
    for (int m = 0; m < KNB; ++m) s += sm.Fs[m * 132 + tid];
    sm.Sred[tid] = sm.sV[tid] + s * (1.0f / SCALEF);
  }
  __syncthreads();

  // ---- LN1, write h1 (f32 + bf16) ----
  if (tid < H) {
    float mu = 0.f;
    for (int i = 0; i < H; ++i) mu += sm.Sred[i];
    mu *= (1.0f / H);
    float var = 0.f;
    for (int i = 0; i < H; ++i) { float d = sm.Sred[i] - mu; var += d * d; }
    var *= (1.0f / H);
    float v = (sm.Sred[tid] - mu) * rsqrtf(var + EPSF) * g1[tid] + be1[tid];
    h1f[(size_t)node * H + tid] = v;
    h1b[(size_t)node * H + tid] = f2bf(v);
  }
}

// ---------------------------------------------------------------------------
// FFN + LN2 + mask_V, 16 nodes per workgroup.
// ---------------------------------------------------------------------------
struct __align__(16) SMemC {
  unsigned short Hm[16 * 520];   // bf16 GELU(Win h) [16 x 512]
  float H2[16 * 132];            // h1 + Wout(...) [16 x 128]
  float mu[16];
  float rs[16];
};

__global__ void __launch_bounds__(256)
ffn_kernel(const float* __restrict__ h1f, const unsigned short* __restrict__ h1b,
           const unsigned short* __restrict__ Winf,  const float* __restrict__ bin,
           const unsigned short* __restrict__ Woutf, const float* __restrict__ bout,
           const float* __restrict__ g2, const float* __restrict__ be2,
           const float* __restrict__ maskV, float* __restrict__ out)
{
  __shared__ SMemC sm;
  const int tid  = threadIdx.x;
  const int lane = tid & 31;
  const int w    = tid >> 5;
  const int base = blockIdx.x * 16;
  const int col  = lane & 15;
  const int half = lane >> 4;

  // GEMM1: h1[16,128] @ Win^T -> GELU -> Hm
  for (int i = 0; i < 4; ++i) {
    int nt = w * 4 + i;                       // 0..31
    f32x8 c = {};
    const unsigned short* arow = h1b + (size_t)(base + col) * H;
    for (int kc = 0; kc < 4; ++kc) {
      FragAB a, b;
      int ab = kc * 32 + half * 8;
      a.q[0] = *(const uint4*)(arow + ab);
      a.q[1] = *(const uint4*)(arow + ab + 16);
      const unsigned short* bp = Winf + (((kc * 32 + nt) * 32 + lane) << 4);
      b.q[0] = *(const uint4*)(bp);
      b.q[1] = *(const uint4*)(bp + 8);
      c = __builtin_amdgcn_wmma_f32_16x16x32_bf16(false, a.v, false, b.v, (short)0, c, false, false);
    }
    float bias = bin[nt * 16 + col];
#pragma unroll
    for (int r = 0; r < 8; ++r) {
      int m = half * 8 + r;
      sm.Hm[m * 520 + nt * 16 + col] = f2bf(gelu_erf(c[r] + bias));
    }
  }
  __syncthreads();

  // GEMM2: Hm[16,512] @ Wout^T + bias + residual -> H2
  {
    int nt = w;                               // 0..7
    f32x8 c = {};
    for (int kc = 0; kc < 16; ++kc) {
      FragAB a, b;
      int abase = col * 520 + kc * 32 + half * 8;
      a.q[0] = *(const uint4*)(&sm.Hm[abase]);
      a.q[1] = *(const uint4*)(&sm.Hm[abase + 16]);
      const unsigned short* bp = Woutf + (((kc * 8 + nt) * 32 + lane) << 4);
      b.q[0] = *(const uint4*)(bp);
      b.q[1] = *(const uint4*)(bp + 8);
      c = __builtin_amdgcn_wmma_f32_16x16x32_bf16(false, a.v, false, b.v, (short)0, c, false, false);
    }
    int n = nt * 16 + col;
    float bias = bout[n];
#pragma unroll
    for (int r = 0; r < 8; ++r) {
      int m = half * 8 + r;
      sm.H2[m * 132 + n] = c[r] + bias + h1f[(size_t)(base + m) * H + n];
    }
  }
  __syncthreads();

  if (tid < 16) {
    float mu = 0.f;
    for (int i = 0; i < H; ++i) mu += sm.H2[tid * 132 + i];
    mu *= (1.0f / H);
    float var = 0.f;
    for (int i = 0; i < H; ++i) { float d = sm.H2[tid * 132 + i] - mu; var += d * d; }
    var *= (1.0f / H);
    sm.mu[tid] = mu;
    sm.rs[tid] = rsqrtf(var + EPSF);
  }
  __syncthreads();

  for (int e = tid; e < 16 * H; e += 256) {
    int m = e >> 7;
    int n = e & 127;
    float v = (sm.H2[m * 132 + n] - sm.mu[m]) * sm.rs[m] * g2[n] + be2[n];
    out[(size_t)(base + m) * H + n] = maskV[base + m] * v;
  }
}

// ---------------------------------------------------------------------------
extern "C" void kernel_launch(void* const* d_in, const int* in_sizes, int n_in,
                              void* d_out, int out_size, void* d_ws, size_t ws_size,
                              hipStream_t stream) {
  const float* hV    = (const float*)d_in[0];
  const float* hE    = (const float*)d_in[1];
  const float* maskV = (const float*)d_in[2];
  const float* maskA = (const float*)d_in[3];
  const float* W1w   = (const float*)d_in[4];
  const float* W1b   = (const float*)d_in[5];
  const float* W2w   = (const float*)d_in[6];
  const float* W2b   = (const float*)d_in[7];
  const float* W3w   = (const float*)d_in[8];
  const float* W3b   = (const float*)d_in[9];
  const float* ln1g  = (const float*)d_in[10];
  const float* ln1b  = (const float*)d_in[11];
  const float* ln2g  = (const float*)d_in[12];
  const float* ln2b  = (const float*)d_in[13];
  const float* Winw  = (const float*)d_in[14];
  const float* Winb  = (const float*)d_in[15];
  const float* Woutw = (const float*)d_in[16];
  const float* Woutb = (const float*)d_in[17];

  char* ws = (char*)d_ws;
  size_t off = 0;
  auto alloc = [&](size_t bytes) {
    void* p = ws + off;
    off = (off + bytes + 255) & ~(size_t)255;
    return p;
  };
  unsigned short* W1f   = (unsigned short*)alloc((size_t)16 * 8  * 512 * 2);
  unsigned short* W2f   = (unsigned short*)alloc((size_t)4  * 8  * 512 * 2);
  unsigned short* W3f   = (unsigned short*)alloc((size_t)4  * 8  * 512 * 2);
  unsigned short* Winf  = (unsigned short*)alloc((size_t)4  * 32 * 512 * 2);
  unsigned short* Woutf = (unsigned short*)alloc((size_t)16 * 8  * 512 * 2);
  float*          h1f   = (float*)         alloc((size_t)NNODES * H * 4);
  unsigned short* h1b   = (unsigned short*)alloc((size_t)NNODES * H * 2);

  auto launch_prep = [&](const float* W, unsigned short* o, int Kdim, int NT, int KC) {
    int total = KC * NT * 512;
    prep_frag<<<(total + 255) / 256, 256, 0, stream>>>(W, o, Kdim, NT, total);
  };
  launch_prep(W1w,   W1f,   512, 8,  16);
  launch_prep(W2w,   W2f,   128, 8,  4);
  launch_prep(W3w,   W3f,   128, 8,  4);
  launch_prep(Winw,  Winf,  128, 32, 4);
  launch_prep(Woutw, Woutf, 512, 8,  16);

  msg_kernel<<<NNODES, 256, 0, stream>>>(hV, hE, maskA,
                                         W1f, W1b, W2f, W2b, W3f, W3b,
                                         ln1g, ln1b, h1f, h1b);
  ffn_kernel<<<NNODES / 16, 256, 0, stream>>>(h1f, h1b, Winf, Winb, Woutf, Woutb,
                                              ln2g, ln2b, maskV, (float*)d_out);
}
